// DART_Net_21603685499652
// MI455X (gfx1250) — matrix operand order repaired and checked by reading, not compiled, not run on hardware
//
#include <hip/hip_runtime.h>

// Problem constants (from the reference)
#define BN   8192          // B*N = 64*128
#define LH   128
#define LO   128
#define GRP  4             // (b,n) groups per block
#define ROWS (GRP * 64)    // 256 rows per block
#define NTHR 256           // 8 waves of 32

typedef float  v8f   __attribute__((ext_vector_type(8)));
typedef __bf16 v16bf __attribute__((ext_vector_type(16)));
typedef __bf16 v8bf  __attribute__((ext_vector_type(8)));

union ABfrag { v16bf v; v8bf h[2]; };

__device__ __forceinline__ float celu01(float x) {
    // jax.nn.celu(x, alpha=0.1) = x>0 ? x : 0.1*(exp(x/0.1)-1)
    return x > 0.f ? x : 0.1f * (__expf(x * 10.f) - 1.f);
}

__device__ __forceinline__ unsigned short f32_to_bf16(float f) {
    unsigned int u = __float_as_uint(f);
    unsigned int r = u + 0x7fffu + ((u >> 16) & 1u);   // round-to-nearest-even
    return (unsigned short)(r >> 16);
}

// one-time f32 -> bf16 weight conversion into workspace
__global__ void cvt_bf16_kernel(const float* __restrict__ src,
                                unsigned short* __restrict__ dst, int n) {
    const int i = blockIdx.x * blockDim.x + threadIdx.x;
    if (i < n) dst[i] = f32_to_bf16(src[i]);
}

__global__ void zero_kernel(float* __restrict__ p, int n) {
    const int i = blockIdx.x * blockDim.x + threadIdx.x;
    if (i < n) p[i] = 0.f;
}

// ---------------------------------------------------------------------------
// mlp2_kernel: 256 rows per block (4 bn-groups):  x[3] -> celu -> 128 -> celu -> 128
//   SUM_OVER_ROWS = true : pair species; masked row-reduction per bn-group,
//                          one global atomic add of the 128-vector per group.
//   SUM_OVER_ROWS = false: "i" species; add each masked row into atm[row].
// Layer 1 (K=3) on VALU; layer 2 (128x128) via v_wmma_f32_16x16x32_bf16.
// 8 waves; wave w owns M-tiles {2w, 2w+1}; B fragments shared across the pair.
// ---------------------------------------------------------------------------
template <bool SUM_OVER_ROWS>
__global__ __launch_bounds__(NTHR) void mlp2_kernel(
    const float* __restrict__ x,                           // [blocks*256][3]
    const float* __restrict__ W1, const float* __restrict__ b1,  // [128][3],[128]
    const unsigned short* __restrict__ W2bf,               // [128][128] bf16 ([n][k])
    const float* __restrict__ b2,                          // [128]
    float* __restrict__ atm)                               // [BN][128] accumulator
{
    __shared__ __align__(32) unsigned short sW2[LO * LH];   // 32 KB
    __shared__ __align__(32) unsigned short sH [ROWS * LH]; // 64 KB
    __shared__ float4 sW1p[LH];                             // (w0,w1,w2,b) per unit
    __shared__ float  sB2[LO];
    __shared__ float  sMask[ROWS];
    __shared__ float  sRed[GRP * LO];

    const int t = threadIdx.x;
    const long long row_base = (long long)blockIdx.x * ROWS;

    // ---- stage bf16 W2 with 128-bit copies; pack layer-1 weights ----
    {
        const uint4* src = reinterpret_cast<const uint4*>(W2bf);
        uint4*       dst = reinterpret_cast<uint4*>(sW2);
        #pragma unroll 4
        for (int idx = t; idx < (LO * LH) / 8; idx += NTHR) dst[idx] = src[idx];
    }
    if (t < LH) sW1p[t] = make_float4(W1[t * 3 + 0], W1[t * 3 + 1], W1[t * 3 + 2], b1[t]);
    if (t < LO) sB2[t] = b2[t];
    for (int idx = t; idx < GRP * LO; idx += NTHR) sRed[idx] = 0.f;
    __syncthreads();

    // ---- layer 1: one thread per row, all 128 hidden units ----
    {
        const float* xr = x + (row_base + t) * 3;
        const float x0 = xr[0], x1 = xr[1], x2 = xr[2];
        sMask[t] = ((x0 + x1 + x2) != 0.f) ? 1.f : 0.f;
        #pragma unroll 8
        for (int u = 0; u < LH; ++u) {
            const float4 w = sW1p[u];
            const float h = fmaf(w.x, x0, fmaf(w.y, x1, fmaf(w.z, x2, w.w)));
            sH[t * LH + u] = f32_to_bf16(celu01(h));
        }
    }
    __syncthreads();

    // ---- layer 2: WMMA 16x16x32 bf16, f32 accumulate ----
    const int lane = t & 31;
    const int ml   = lane & 15;     // row (A/C) or col (B/C) within tile
    const int g    = lane >> 4;     // K half selector
    const int wv   = t >> 5;        // wave id (0..7)
    const int mt0  = wv * 2;        // first of two M-tiles for this wave
    const int grp  = wv >> 1;       // bn-group (both tiles share it)

    // A fragments (both M-tiles, all 4 K-blocks), reused across 8 N-tiles.
    // A layout: lanes 0-15/16-31 carry K = g*16..+15 (contiguous, matches B).
    ABfrag A[2][4];
    #pragma unroll
    for (int i = 0; i < 2; ++i)
        #pragma unroll
        for (int kb = 0; kb < 4; ++kb) {
            const int base = ((mt0 + i) * 16 + ml) * LH + kb * 32 + g * 16;
            A[i][kb].h[0] = *reinterpret_cast<const v8bf*>(&sH[base]);
            A[i][kb].h[1] = *reinterpret_cast<const v8bf*>(&sH[base + 8]);
        }

    // C layout: VGPR v holds row M = v + 8*g within the tile
    float mrow[2][8];
    #pragma unroll
    for (int i = 0; i < 2; ++i)
        #pragma unroll
        for (int v = 0; v < 8; ++v) mrow[i][v] = sMask[(mt0 + i) * 16 + g * 8 + v];

    #pragma unroll
    for (int nt = 0; nt < 8; ++nt) {
        // bias is column-constant -> fold it into the accumulator init
        const float bias = sB2[nt * 16 + ml];
        v8f c0 = {bias, bias, bias, bias, bias, bias, bias, bias};
        v8f c1 = c0;
        #pragma unroll
        for (int kb = 0; kb < 4; ++kb) {
            const v16bf bf = *reinterpret_cast<const v16bf*>(
                &sW2[(nt * 16 + ml) * LH + kb * 32 + g * 16]);
            c0 = __builtin_amdgcn_wmma_f32_16x16x32_bf16(
                     false, A[0][kb].v, false, bf, (short)0, c0, false, false);
            c1 = __builtin_amdgcn_wmma_f32_16x16x32_bf16(
                     false, A[1][kb].v, false, bf, (short)0, c1, false, false);
        }
        if (SUM_OVER_ROWS) {
            float s0 = 0.f, s1 = 0.f;
            #pragma unroll
            for (int v = 0; v < 8; ++v) {
                s0 += celu01(c0[v]) * mrow[0][v];
                s1 += celu01(c1[v]) * mrow[1][v];
            }
            s0 += __shfl_xor(s0, 16, 32);       // combine the two row halves
            s1 += __shfl_xor(s1, 16, 32);
            if (lane < 16) {
                atomicAdd(&sRed[grp * LO + nt * 16 + ml], s0 + s1);
            }
        } else {
            #pragma unroll
            for (int v = 0; v < 8; ++v) {
                const float val0 = celu01(c0[v]) * mrow[0][v];
                const float val1 = celu01(c1[v]) * mrow[1][v];
                const long long r0 = row_base + (mt0 + 0) * 16 + g * 8 + v;
                const long long r1 = row_base + (mt0 + 1) * 16 + g * 8 + v;
                atm[r0 * LO + nt * 16 + ml] += val0;   // unique (row,col) per lane
                atm[r1 * LO + nt * 16 + ml] += val1;
            }
        }
    }

    if (SUM_OVER_ROWS) {
        __syncthreads();
        for (int idx = t; idx < GRP * LO; idx += NTHR) {
            const long long bn = (long long)blockIdx.x * GRP + (idx >> 7);
            atomicAdd(&atm[bn * LO + (idx & 127)], sRed[idx]);
        }
    }
}

// ---------------------------------------------------------------------------
// head_kernel: one block per (b,n) row: 128 ->64 ->32 ->16 ->1, times ai_mask
// ---------------------------------------------------------------------------
__global__ __launch_bounds__(128) void head_kernel(
    const float* __restrict__ atm, const float* __restrict__ ai,
    const float* __restrict__ W1, const float* __restrict__ b1,
    const float* __restrict__ W2, const float* __restrict__ b2,
    const float* __restrict__ W3, const float* __restrict__ b3,
    const float* __restrict__ W4, const float* __restrict__ b4,
    float* __restrict__ out)
{
    __shared__ float sRow[128];
    __shared__ float sH1[64];
    __shared__ float sH2[32];
    __shared__ float sH3[16];
    const int t = threadIdx.x;
    const long long row = blockIdx.x;

    sRow[t] = atm[row * 128 + t];
    __syncthreads();
    if (t < 64) {
        float acc = b1[t];
        #pragma unroll 16
        for (int k = 0; k < 128; ++k) acc += W1[t * 128 + k] * sRow[k];
        sH1[t] = celu01(acc);
    }
    __syncthreads();
    if (t < 32) {
        float acc = b2[t];
        #pragma unroll 16
        for (int k = 0; k < 64; ++k) acc += W2[t * 64 + k] * sH1[k];
        sH2[t] = celu01(acc);
    }
    __syncthreads();
    if (t < 16) {
        float acc = b3[t];
        #pragma unroll
        for (int k = 0; k < 32; ++k) acc += W3[t * 32 + k] * sH2[k];
        sH3[t] = celu01(acc);
    }
    __syncthreads();
    if (t == 0) {
        float acc = b4[0];
        #pragma unroll
        for (int k = 0; k < 16; ++k) acc += W4[k] * sH3[k];
        const float s = ai[row * 3 + 0] + ai[row * 3 + 1] + ai[row * 3 + 2];
        out[row] = (s != 0.f) ? acc : 0.f;    // final * ai_mask
    }
}

// ---------------------------------------------------------------------------
extern "C" void kernel_launch(void* const* d_in, const int* in_sizes, int n_in,
                              void* d_out, int out_size, void* d_ws, size_t ws_size,
                              hipStream_t stream)
{
    (void)in_sizes; (void)n_in; (void)out_size; (void)ws_size;

    const float* ai  = (const float*)d_in[0];
    const float* aj  = (const float*)d_in[1];
    const float* ak  = (const float*)d_in[2];
    const float* al  = (const float*)d_in[3];
    const float* Wi1 = (const float*)d_in[4];  const float* bi1 = (const float*)d_in[5];
    const float* Wi2 = (const float*)d_in[6];  const float* bi2 = (const float*)d_in[7];
    const float* Wj1 = (const float*)d_in[8];  const float* bj1 = (const float*)d_in[9];
    const float* Wj2 = (const float*)d_in[10]; const float* bj2 = (const float*)d_in[11];
    const float* Wk1 = (const float*)d_in[12]; const float* bk1 = (const float*)d_in[13];
    const float* Wk2 = (const float*)d_in[14]; const float* bk2 = (const float*)d_in[15];
    const float* Wl1 = (const float*)d_in[16]; const float* bl1 = (const float*)d_in[17];
    const float* Wl2 = (const float*)d_in[18]; const float* bl2 = (const float*)d_in[19];
    const float* W1  = (const float*)d_in[20]; const float* b1  = (const float*)d_in[21];
    const float* W2  = (const float*)d_in[22]; const float* b2  = (const float*)d_in[23];
    const float* W3  = (const float*)d_in[24]; const float* b3  = (const float*)d_in[25];
    const float* W4  = (const float*)d_in[26]; const float* b4  = (const float*)d_in[27];

    // workspace layout: atm [BN][128] f32 (4 MB), then 4x bf16 W2 (32 KB each)
    float* atm = (float*)d_ws;
    unsigned short* w2base =
        (unsigned short*)((char*)d_ws + (size_t)BN * LO * sizeof(float));
    unsigned short* w2i = w2base;
    unsigned short* w2j = w2base + 1 * LO * LH;
    unsigned short* w2k = w2base + 2 * LO * LH;
    unsigned short* w2l = w2base + 3 * LO * LH;
    float* out = (float*)d_out;

    const int wn = LO * LH;                          // 16384
    cvt_bf16_kernel<<<(wn + 255) / 256, 256, 0, stream>>>(Wi2, w2i, wn);
    cvt_bf16_kernel<<<(wn + 255) / 256, 256, 0, stream>>>(Wj2, w2j, wn);
    cvt_bf16_kernel<<<(wn + 255) / 256, 256, 0, stream>>>(Wk2, w2k, wn);
    cvt_bf16_kernel<<<(wn + 255) / 256, 256, 0, stream>>>(Wl2, w2l, wn);

    const int atm_n = BN * LO;
    zero_kernel<<<(atm_n + 255) / 256, 256, 0, stream>>>(atm, atm_n);

    // pair species: 256 rows (4 bn-groups) per block, reduce over M
    mlp2_kernel<true ><<<BN / GRP, NTHR, 0, stream>>>(aj, Wj1, bj1, w2j, bj2, atm);
    mlp2_kernel<true ><<<BN / GRP, NTHR, 0, stream>>>(ak, Wk1, bk1, w2k, bk2, atm);
    mlp2_kernel<true ><<<BN / GRP, NTHR, 0, stream>>>(al, Wl1, bl1, w2l, bl2, atm);

    // self species: 8192 rows total, 256 rows per block, add per-row
    mlp2_kernel<false><<<BN / ROWS, NTHR, 0, stream>>>(ai, Wi1, bi1, w2i, bi2, atm);

    // readout head
    head_kernel<<<BN, 128, 0, stream>>>(atm, ai, W1, b1, W2, b2, W3, b3, W4, b4, out);
}
